// VectorQuantStraightThrough_7679401525803
// MI455X (gfx1250) — matrix operand order repaired and checked by reading, not compiled, not run on hardware
//
#include <hip/hip_runtime.h>

// ---------------------------------------------------------------------------
// Vector-quantize (VQ-VAE straight-through) for MI455X / gfx1250.
//   inputs : z_e [32,4096,64] f32, emb [512,64] f32
//   outputs: z_q_st [B,V,D] f32, z_q [B,V,D] f32, indices [B,V] (as f32)
// GEMM z @ emb^T via v_wmma_f32_16x16x32_f16, fused online argmin.
// ---------------------------------------------------------------------------

typedef __attribute__((ext_vector_type(16))) _Float16 v16h;
typedef __attribute__((ext_vector_type(8)))  float    v8f;

#define N_ROWS      (32 * 4096)   // 131072 flattened rows
#define KCODES      512
#define DDIM        64
#define ROW_STRIDE  80            // padded halfs per code row in LDS (160 B, 32B-aligned sublanes)
#define WG_ROWS     128           // rows per workgroup (8 waves x 16 rows)

__global__ __launch_bounds__(256)
void vq_argmin_kernel(const float* __restrict__ z_e,
                      const float* __restrict__ emb,
                      float* __restrict__ out)
{
    extern __shared__ v16h smem_v[];                       // 32B-aligned LDS base
    _Float16* embLds  = (_Float16*)smem_v;                 // [512 * 80] f16
    float*    emb2Lds = (float*)(embLds + KCODES * ROW_STRIDE); // [512] f32

    const int tid = threadIdx.x;

    // ---- stage codebook into LDS as f16 (padded row-major) -----------------
    for (int i = tid; i < KCODES * DDIM; i += 256) {
        const int r = i >> 6, c = i & 63;
        embLds[r * ROW_STRIDE + c] = (_Float16)emb[i];
    }
    // ---- per-code squared norms (f32, exact) -------------------------------
    for (int k = tid; k < KCODES; k += 256) {
        const float* e = emb + k * DDIM;
        float s = 0.f;
        #pragma unroll
        for (int d = 0; d < DDIM; ++d) s = __builtin_fmaf(e[d], e[d], s);
        emb2Lds[k] = s;
    }
    __syncthreads();

    const int wave = tid >> 5;
    const int lane = tid & 31;
    const int hi   = lane >> 4;      // half-wave select
    const int nl   = lane & 15;
    const int row_base = blockIdx.x * WG_ROWS + wave * 16;

    // ---- A operand: 16 rows x 64 k, f32 -> f16, contiguous k = 16*hi + s --------
    // lane (nl) holds row M = nl; lo half has K 0..15+kb, hi half 16..31+kb
    const float* zr = z_e + (size_t)(row_base + nl) * DDIM + 16 * hi;
    v16h a0, a1;
    #pragma unroll
    for (int j = 0; j < 8; ++j) {
        a0[j]     = (_Float16)zr[j];        // kb=0,  K = 16*hi + j
        a0[8 + j] = (_Float16)zr[8 + j];    // kb=0,  K = 16*hi + 8 + j
        a1[j]     = (_Float16)zr[32 + j];   // kb=32
        a1[8 + j] = (_Float16)zr[40 + j];
    }

    float minv[8];
    int   mini[8];
    #pragma unroll
    for (int r = 0; r < 8; ++r) { minv[r] = 3.4e38f; mini[r] = 0; }

    // ---- sweep 32 column tiles of the codebook -----------------------------
    #pragma unroll 4
    for (int t = 0; t < KCODES / 16; ++t) {
        const int n = t * 16 + nl;                         // this lane's column (code id)
        // B operand (32x16 f16): lane column N=nl, lo half K=0..15, hi half K=16..31 (+kb)
        const _Float16* bp = embLds + n * ROW_STRIDE + 16 * hi;
        const v16h b0 = *(const v16h*)(bp);                // kb = 0
        const v16h b1 = *(const v16h*)(bp + 32);           // kb = 32

        v8f acc = {};
        acc = __builtin_amdgcn_wmma_f32_16x16x32_f16(false, a0, false, b0,
                                                     (short)0, acc, false, false);
        acc = __builtin_amdgcn_wmma_f32_16x16x32_f16(false, a1, false, b1,
                                                     (short)0, acc, false, false);

        // fused online argmin:  score = ||e_n||^2 - 2 * (z . e_n)
        const float e2 = emb2Lds[n];
        #pragma unroll
        for (int r = 0; r < 8; ++r) {                      // C layout: row M = r + 8*hi
            const float sc = __builtin_fmaf(-2.f, acc[r], e2);
            if (sc < minv[r]) { minv[r] = sc; mini[r] = n; }
        }
    }

    // ---- cross-lane min+index reduction over the 16 column lanes -----------
    #pragma unroll
    for (int off = 1; off < 16; off <<= 1) {
        #pragma unroll
        for (int r = 0; r < 8; ++r) {
            const float ov = __shfl_xor(minv[r], off, 32);
            const int   oi = __shfl_xor(mini[r], off, 32);
            if (ov < minv[r] || (ov == minv[r] && oi < mini[r])) {
                minv[r] = ov; mini[r] = oi;                // first-min tie-break
            }
        }
    }

    const size_t O2 = (size_t)N_ROWS * DDIM;   // z_q block offset
    const size_t O3 = 2 * O2;                  // indices block offset

    // indices (stored as float): lanes 0 and 16 cover rows 0..7 / 8..15
    if (nl == 0) {
        #pragma unroll
        for (int r = 0; r < 8; ++r)
            out[O3 + (size_t)(row_base + 8 * hi + r)] = (float)mini[r];
    }

    // ---- gather selected code rows into z_q_st and z_q (identical fwd) -----
    #pragma unroll 2
    for (int m = 0; m < 16; ++m) {
        const int idx = __shfl(mini[m & 7], (m >> 3) * 16, 32);  // broadcast row m's argmin
        const float* src = emb + (size_t)idx * DDIM;             // L2-resident
        const float v0 = src[lane];
        const float v1 = src[lane + 32];
        float* o = out + (size_t)(row_base + m) * DDIM;
        o[lane]          = v0;  o[lane + 32]      = v1;          // z_q_st
        o[O2 + lane]     = v0;  o[O2 + lane + 32] = v1;          // z_q
    }
}

extern "C" void kernel_launch(void* const* d_in, const int* in_sizes, int n_in,
                              void* d_out, int out_size, void* d_ws, size_t ws_size,
                              hipStream_t stream) {
    (void)in_sizes; (void)n_in; (void)out_size; (void)d_ws; (void)ws_size;
    const float* z_e = (const float*)d_in[0];
    const float* emb = (const float*)d_in[1];
    float* out = (float*)d_out;

    const size_t shmem = (size_t)KCODES * ROW_STRIDE * sizeof(_Float16)  // 81920 B
                       + (size_t)KCODES * sizeof(float);                 // + 2048 B
    dim3 grid(N_ROWS / WG_ROWS);   // 1024 workgroups
    dim3 block(256);               // 8 wave32s
    hipLaunchKernelGGL(vq_argmin_kernel, grid, block, shmem, stream, z_e, emb, out);
}
